// MultiHeadedAttention_45105746542633
// MI455X (gfx1250) — matrix operand order repaired and checked by reading, not compiled, not run on hardware
//
#include <hip/hip_runtime.h>

typedef __attribute__((ext_vector_type(16))) _Float16 v16h;
typedef __attribute__((ext_vector_type(8)))  _Float16 v8h;
typedef __attribute__((ext_vector_type(8)))  float    v8f;
typedef __attribute__((ext_vector_type(4)))  float    v4f;
typedef __attribute__((ext_vector_type(4)))  unsigned int u32x4;
typedef __attribute__((ext_vector_type(8)))  int      i32x8;
typedef __attribute__((ext_vector_type(4)))  int      i32x4;

#define NTOK   3072
#define DMODEL 256
#define NHEAD  8
#define DK     32

static __device__ __forceinline__ v8f wmma_f16(v16h a, v16h b, v8f c) {
  return __builtin_amdgcn_wmma_f32_16x16x32_f16(false, a, false, b, (short)0, c, false, false);
}

// A-operand (16x32 f16) from f16 memory: per lane one contiguous 16-half run.
static __device__ __forceinline__ v16h loadA16(const _Float16* p) {
  v8h lo = *(const v8h*)p;
  v8h hi = *(const v8h*)(p + 8);
  v16h a;
#pragma unroll
  for (int e = 0; e < 8; ++e) { a[e] = lo[e]; a[8 + e] = hi[e]; }
  return a;
}

// A-operand from f32 memory (contiguous run at +0..15), converted to f16.
static __device__ __forceinline__ v16h loadA32(const float* p) {
  v4f x0 = *(const v4f*)p;
  v4f x1 = *(const v4f*)(p + 4);
  v4f x2 = *(const v4f*)(p + 8);
  v4f x3 = *(const v4f*)(p + 12);
  v16h a;
#pragma unroll
  for (int e = 0; e < 4; ++e) {
    a[e]      = (_Float16)x0[e];
    a[4 + e]  = (_Float16)x1[e];
    a[8 + e]  = (_Float16)x2[e];
    a[12 + e] = (_Float16)x3[e];
  }
  return a;
}

// B-operand from 16 contiguous f32 (global or LDS), converted to f16 * scale.
static __device__ __forceinline__ v16h loadB32(const float* p, float scale) {
  v4f x0 = *(const v4f*)p;
  v4f x1 = *(const v4f*)(p + 4);
  v4f x2 = *(const v4f*)(p + 8);
  v4f x3 = *(const v4f*)(p + 12);
  v16h b;
#pragma unroll
  for (int e = 0; e < 4; ++e) {
    b[e]      = (_Float16)(x0[e] * scale);
    b[4 + e]  = (_Float16)(x1[e] * scale);
    b[8 + e]  = (_Float16)(x2[e] * scale);
    b[12 + e] = (_Float16)(x3[e] * scale);
  }
  return b;
}

// ---------------------------------------------------------------------------
// Tensor Data Mover: async 2-D f32 tile (tile_w x tile_h) global -> LDS.
// D# per CDNA5 ISA ch.8: group0 = {count, lds_addr, global_addr(57b), type=2},
// group1 = {data_size, tensor dims, tile dims, dim0 stride}. Groups 2/3 zero
// (2-D tile, <=2 descriptor groups needed).
// ---------------------------------------------------------------------------
static __device__ __forceinline__ void tdm_load_tile_f32(
    unsigned lds_byte_addr, const float* tile_origin, unsigned tensor_w,
    unsigned tensor_h, unsigned row_stride_elems, unsigned tile_w, unsigned tile_h) {
  const unsigned long long ga = (unsigned long long)(uintptr_t)tile_origin;
  u32x4 g0;
  g0[0] = 1u;                                    // count=1, user descriptor
  g0[1] = lds_byte_addr;                         // lds_addr [63:32]
  g0[2] = (unsigned)(ga & 0xffffffffu);          // global_addr[31:0]
  g0[3] = (unsigned)((ga >> 32) & 0x01ffffffu)   // global_addr[56:32]
        | (2u << 30);                            // type = 2 ("image")
  i32x8 g1;
  g1[0] = (int)(2u << 16);                                         // data_size=4B
  g1[1] = (int)((tensor_w & 0xffffu) << 16);                       // dim0[15:0] @bit48
  g1[2] = (int)((tensor_w >> 16) | ((tensor_h & 0xffffu) << 16));  // dim0[31:16],dim1[15:0]
  g1[3] = (int)((tensor_h >> 16) | (tile_w << 16));                // dim1[31:16],tile_dim0
  g1[4] = (int)tile_h;                                             // tile_dim1 (tile_dim2=0)
  g1[5] = (int)row_stride_elems;                                   // dim0_stride[31:0]
  g1[6] = 0;
  g1[7] = 0;
  i32x4 z4 = {0, 0, 0, 0};
#if defined(__clang_major__) && (__clang_major__ >= 23)
  i32x8 z8 = {0, 0, 0, 0, 0, 0, 0, 0};
  __builtin_amdgcn_tensor_load_to_lds(g0, g1, z4, z4, z8, 0);
#else
  __builtin_amdgcn_tensor_load_to_lds(g0, g1, z4, z4, 0);
#endif
}

static __device__ __forceinline__ unsigned lds_addr_of(const void* p) {
  return (unsigned)(uintptr_t)p;  // generic LDS pointer: low 32 bits = LDS offset
}

// ---------------------------------------------------------------------------
// 16x16-tile GEMM for C = X[N,256] @ W^T + b, one wave per tile.
// MODE 0: A=W, B=X -> C^T; store f16 [H][N][DK] packed, value*scale. (Q / K)
// MODE 1: A=X, B=W -> C;   store f16 [H][DK][N] packed.              (V^T)
// MODE 2: A=W, B=X -> C^T; store f32 [N,256] contiguous.             (output)
// ---------------------------------------------------------------------------
template <int MODE>
__global__ void gemm16(const float* __restrict__ X, const float* __restrict__ W,
                       const float* __restrict__ bias, void* __restrict__ dstv,
                       float scale) {
  const int lane = threadIdx.x & 31;
  const int half = lane >> 4;
  const int ml   = lane & 15;
  const int m0   = blockIdx.x * 16;  // token tile
  const int n0   = blockIdx.y * 16;  // feature tile

  v8f acc = {};
  if (MODE == 1) {
    const float* arow = X + (size_t)(m0 + ml) * DMODEL;
    const float* brow = W + (size_t)(n0 + ml) * DMODEL;
#pragma unroll
    for (int kc = 0; kc < DMODEL / 32; ++kc) {
      v16h a = loadA32(arow + kc * 32 + half * 16);
      v16h b = loadB32(brow + kc * 32 + half * 16, 1.0f);
      acc = wmma_f16(a, b, acc);
    }
    const float bv = bias[n0 + ml];
    _Float16* dst = (_Float16*)dstv;
    const int h = n0 >> 5, d = (n0 & 31) + ml;
    v8h pk;
#pragma unroll
    for (int j = 0; j < 8; ++j) pk[j] = (_Float16)((acc[j] + bv) * scale);
    *(v8h*)(dst + ((size_t)h * DK + d) * NTOK + m0 + half * 8) = pk;
  } else {
    const float* arow = W + (size_t)(n0 + ml) * DMODEL;
    const float* brow = X + (size_t)(m0 + ml) * DMODEL;
#pragma unroll
    for (int kc = 0; kc < DMODEL / 32; ++kc) {
      v16h a = loadA32(arow + kc * 32 + half * 16);
      v16h b = loadB32(brow + kc * 32 + half * 16, 1.0f);
      acc = wmma_f16(a, b, acc);
    }
    v4f b0 = *(const v4f*)(bias + n0 + half * 8);
    v4f b1 = *(const v4f*)(bias + n0 + half * 8 + 4);
    if (MODE == 0) {
      _Float16* dst = (_Float16*)dstv;
      const int h = n0 >> 5, d0 = n0 & 31;
      v8h pk;
#pragma unroll
      for (int j = 0; j < 8; ++j) {
        const float bv = (j < 4) ? b0[j] : b1[j - 4];
        pk[j] = (_Float16)((acc[j] + bv) * scale);
      }
      *(v8h*)(dst + ((size_t)h * NTOK + m0 + ml) * DK + d0 + half * 8) = pk;
    } else {
      float* dst = (float*)dstv;
      float* orow = dst + (size_t)(m0 + ml) * DMODEL + n0 + half * 8;
      v4f r0, r1;
#pragma unroll
      for (int j = 0; j < 4; ++j) {
        r0[j] = acc[j] + b0[j];
        r1[j] = acc[4 + j] + b1[j];
      }
      *(v4f*)orow = r0;
      *(v4f*)(orow + 4) = r1;
    }
  }
}

// ---------------------------------------------------------------------------
// invL1[r] = lambda1 / (sum_c adj[r][c] + eps)
// ---------------------------------------------------------------------------
__global__ void rowinv(const float* __restrict__ adj, const float* __restrict__ lambdas,
                       float* __restrict__ invL1) {
  const int r = blockIdx.x;
  const int lane = threadIdx.x & 31;
  float s = 0.f;
  for (int c = lane; c < NTOK; c += 32) s += adj[(size_t)r * NTOK + c];
#pragma unroll
  for (int off = 16; off; off >>= 1) s += __shfl_xor(s, off, 32);
  if (lane == 0) invL1[r] = lambdas[1] / (s + 1e-6f);
}

// ---------------------------------------------------------------------------
// Fused attention, transposed-score formulation. One wave per (16 queries, head).
//   S^T = K_tile (A) x Q_tile (B);  O^T = V^T (A) x [P^T | adj^T] (B)
// Per-lane scalar softmax state. adj + mask f32 tiles streamed via the Tensor
// Data Mover into double-buffered LDS, pipelined one chunk ahead
// (tensor_load_to_lds + s_wait_tensorcnt). Writes Xm[N,256] f32, contiguous.
// ---------------------------------------------------------------------------
__global__ void attn_fused(const _Float16* __restrict__ Qs, const _Float16* __restrict__ Ks,
                           const _Float16* __restrict__ Vt, const float* __restrict__ adj,
                           const float* __restrict__ maskp, const float* __restrict__ invL1,
                           const float* __restrict__ lambdas, float* __restrict__ Xm) {
  const int lane = threadIdx.x & 31;
  const int half = lane >> 4;
  const int ml   = lane & 15;
  const int q0   = blockIdx.x * 16;
  const int h    = blockIdx.y;

  const _Float16* Qh = Qs + (size_t)h * NTOK * DK;
  const _Float16* Kh = Ks + (size_t)h * NTOK * DK;
  const _Float16* Vh = Vt + (size_t)h * DK * NTOK;

  __shared__ float    AdjLds[2][16][32];   // TDM-filled adjacency tiles
  __shared__ float    MskLds[2][16][32];   // TDM-filled mask tiles
  __shared__ _Float16 Pb[2][16][32];       // P^T staging [query][key]

  // Q as B-operand (32 d x 16 queries), loaded once. Q pre-scaled by 1/sqrt(dk).
  v16h bq = *(const v16h*)(Qh + (size_t)(q0 + ml) * DK + half * 16);

  const float  invr    = invL1[q0 + ml];                  // lambda1/(rowsum+eps)
  const float* adjbase = adj   + (size_t)q0 * NTOK;       // tile origins (uniform)
  const float* mskbase = maskp + (size_t)q0 * NTOK;

  // prologue: stream chunk 0 tiles into buffer 0
  tdm_load_tile_f32(lds_addr_of(&AdjLds[0][0][0]), adjbase, NTOK, NTOK, NTOK, 32, 16);
  tdm_load_tile_f32(lds_addr_of(&MskLds[0][0][0]), mskbase, NTOK, NTOK, NTOK, 32, 16);

  float mold = -1e30f, lsum = 0.f;
  v8f oT0 = {}, oT1 = {}, oG0 = {}, oG1 = {};

  for (int c0 = 0; c0 < NTOK; c0 += 32) {
    const int  pb   = (c0 >> 5) & 1;
    const bool more = (c0 + 32) < NTOK;
    if (more) {  // stream next chunk into the other buffer while computing
      tdm_load_tile_f32(lds_addr_of(&AdjLds[pb ^ 1][0][0]), adjbase + c0 + 32,
                        NTOK, NTOK, NTOK, 32, 16);
      tdm_load_tile_f32(lds_addr_of(&MskLds[pb ^ 1][0][0]), mskbase + c0 + 32,
                        NTOK, NTOK, NTOK, 32, 16);
    }

    // K tiles as A-operands (keys are rows); overlap with tensorcnt wait
    v16h ak0 = loadA16(Kh + (size_t)(c0 + ml) * DK + half * 16);
    v16h ak1 = loadA16(Kh + (size_t)(c0 + 16 + ml) * DK + half * 16);
    __builtin_prefetch(Kh + (size_t)(c0 + 32 + ml) * DK, 0, 0);

    // current chunk's 2 TDM loads complete (same-wave TDM ops finish in order);
    // s_wait_tensorcnt needs a constant immediate, so branch on the uniform flag
    if (more) {
      __builtin_amdgcn_s_wait_tensorcnt(2);
    } else {
      __builtin_amdgcn_s_wait_tensorcnt(0);
    }
    __asm__ __volatile__("" ::: "memory");

    v8f z = {};
    v8f s0 = wmma_f16(ak0, bq, z);  // S^T keys c0..c0+15
    v8f s1 = wmma_f16(ak1, bq, z);  // S^T keys c0+16..c0+31

    // additive mask from LDS: per lane contiguous along keys (j + 8*half)
    v4f mk0a = *(const v4f*)&MskLds[pb][ml][half * 8];
    v4f mk0b = *(const v4f*)&MskLds[pb][ml][half * 8 + 4];
    v4f mk1a = *(const v4f*)&MskLds[pb][ml][16 + half * 8];
    v4f mk1b = *(const v4f*)&MskLds[pb][ml][16 + half * 8 + 4];
#pragma unroll
    for (int j = 0; j < 4; ++j) {
      s0[j]     += mk0a[j];
      s0[4 + j] += mk0b[j];
      s1[j]     += mk1a[j];
      s1[4 + j] += mk1b[j];
    }

    // online softmax: per-lane scalar state (query = ml)
    float t = -1e30f;
#pragma unroll
    for (int j = 0; j < 8; ++j) t = fmaxf(t, fmaxf(s0[j], s1[j]));
    t = fmaxf(t, __shfl_xor(t, 16, 32));
    const float mnew = fmaxf(mold, t);
    const float al   = __expf(mold - mnew);
    mold = mnew;

    float psum = 0.f;
    v8h p0h, p1h;
#pragma unroll
    for (int j = 0; j < 8; ++j) {
      const float p0 = __expf(s0[j] - mnew);
      const float p1 = __expf(s1[j] - mnew);
      psum += p0 + p1;
      p0h[j] = (_Float16)p0;
      p1h[j] = (_Float16)p1;
    }
    lsum = lsum * al + psum;
#pragma unroll
    for (int j = 0; j < 8; ++j) { oT0[j] *= al; oT1[j] *= al; }

    // stage P^T: [query][key], contiguous in j -> single b128 stores
    *(v8h*)&Pb[pb][ml][half * 8]      = p0h;
    *(v8h*)&Pb[pb][ml][16 + half * 8] = p1h;

    // read back as B-operand (32 keys x 16 queries)
    v16h bp = *(const v16h*)&Pb[pb][ml][half * 16];

    // V^T tiles as A-operands (d rows, contiguous keys)
    v16h av0 = loadA16(Vh + (size_t)ml * NTOK + c0 + half * 16);
    v16h av1 = loadA16(Vh + (size_t)(16 + ml) * NTOK + c0 + half * 16);

    // adjacency as B-operand from LDS, lambda1/rowsum folded in
    v16h ba = loadB32(&AdjLds[pb][ml][half * 16], invr);

    oT0 = wmma_f16(av0, bp, oT0);
    oT1 = wmma_f16(av1, bp, oT1);
    oG0 = wmma_f16(av0, ba, oG0);
    oG1 = wmma_f16(av1, ba, oG1);
  }

  // finalize: lambda0 * O_attn / l + O_adj ; per-lane query column, contiguous d
  const float l  = lsum + __shfl_xor(lsum, 16, 32);
  const float sc = lambdas[0] / l;
  float* orow = Xm + (size_t)(q0 + ml) * DMODEL + h * DK;
  v4f r0, r1, r2, r3;
#pragma unroll
  for (int j = 0; j < 4; ++j) {
    r0[j] = oT0[j] * sc + oG0[j];
    r1[j] = oT0[4 + j] * sc + oG0[4 + j];
    r2[j] = oT1[j] * sc + oG1[j];
    r3[j] = oT1[4 + j] * sc + oG1[4 + j];
  }
  *(v4f*)(orow + half * 8)          = r0;
  *(v4f*)(orow + half * 8 + 4)      = r1;
  *(v4f*)(orow + 16 + half * 8)     = r2;
  *(v4f*)(orow + 16 + half * 8 + 4) = r3;
}

// ---------------------------------------------------------------------------
extern "C" void kernel_launch(void* const* d_in, const int* in_sizes, int n_in,
                              void* d_out, int out_size, void* d_ws, size_t ws_size,
                              hipStream_t stream) {
  const float* query   = (const float*)d_in[0];
  const float* key     = (const float*)d_in[1];
  const float* value   = (const float*)d_in[2];
  const float* maskp   = (const float*)d_in[3];
  const float* adj     = (const float*)d_in[4];
  const float* lambdas = (const float*)d_in[5];
  const float* Wq = (const float*)d_in[6];
  const float* bq = (const float*)d_in[7];
  const float* Wk = (const float*)d_in[8];
  const float* bk = (const float*)d_in[9];
  const float* Wv = (const float*)d_in[10];
  const float* bv = (const float*)d_in[11];
  const float* Wo = (const float*)d_in[12];
  const float* bo = (const float*)d_in[13];

  char* ws = (char*)d_ws;
  size_t off = 0;
  auto carve = [&](size_t bytes) {
    void* p = ws + off;
    off = (off + bytes + 255) & ~(size_t)255;
    return p;
  };
  const size_t qkv_bytes = (size_t)NHEAD * NTOK * DK * sizeof(_Float16);
  _Float16* Qs   = (_Float16*)carve(qkv_bytes);
  _Float16* Ks   = (_Float16*)carve(qkv_bytes);
  _Float16* Vt   = (_Float16*)carve(qkv_bytes);
  float*    invL = (float*)carve((size_t)NTOK * sizeof(float));
  float*    Xm   = (float*)carve((size_t)NTOK * DMODEL * sizeof(float));

  const dim3 gG(NTOK / 16, DMODEL / 16);
  const float qscale = 0.17677669529663687f;  // 1/sqrt(32)

  gemm16<0><<<gG, 32, 0, stream>>>(query, Wq, bq, (void*)Qs, qscale);
  gemm16<0><<<gG, 32, 0, stream>>>(key,   Wk, bk, (void*)Ks, 1.0f);
  gemm16<1><<<gG, 32, 0, stream>>>(value, Wv, bv, (void*)Vt, 1.0f);
  rowinv<<<NTOK, 32, 0, stream>>>(adj, lambdas, invL);

  attn_fused<<<dim3(NTOK / 16, NHEAD), 32, 0, stream>>>(Qs, Ks, Vt, adj, maskp,
                                                        invL, lambdas, Xm);

  gemm16<2><<<gG, 32, 0, stream>>>(Xm, Wo, bo, d_out, 1.0f);
}
